// GNN_82008105550480
// MI455X (gfx1250) — matrix operand order repaired and checked by reading, not compiled, not run on hardware
//
#include <hip/hip_runtime.h>
#include <hip/hip_bf16.h>

#define NN 65536
#define NE 262144
#define TW 25
#define NV 4
#define EMB 128
#define LAYERS 6
#define KP 288            // padded K for the W1 GEMMs (286 msg / 260 upd -> 288 = 9*32)
#define T_MAXF 16.0f

typedef _Float16 v16h __attribute__((ext_vector_type(16)));
typedef _Float16 v8h  __attribute__((ext_vector_type(8)));
typedef float    v8f  __attribute__((ext_vector_type(8)));

#if defined(__AMDGCN__) && __has_builtin(__builtin_amdgcn_tensor_load_to_lds)
#define HAVE_TDM 1
typedef unsigned int u32x4 __attribute__((ext_vector_type(4)));
typedef int          i32x8 __attribute__((ext_vector_type(8)));
typedef int          i32x4 __attribute__((ext_vector_type(4)));
#endif

__device__ __forceinline__ float fast_rcp(float x) {
#if defined(__AMDGCN__)
  return __builtin_amdgcn_rcpf(x);
#else
  return 1.f / x;
#endif
}
__device__ __forceinline__ float fast_rsq(float x) {
#if defined(__AMDGCN__)
  return __builtin_amdgcn_rsqf(x);
#else
  return rsqrtf(x);
#endif
}
__device__ __forceinline__ float silu_f(float x) {
  return x * fast_rcp(1.f + __expf(-x));   // v_exp + v_add + v_rcp + v_mul (no IEEE div chain)
}

__device__ __forceinline__ v8f wmma16(v16h a, v16h b, v8f c) {
  return __builtin_amdgcn_wmma_f32_16x16x32_f16(false, a, false, b, (short)0, c, false, false);
}

// A fragment: 16xK row-major tile in LDS (stride in halves).
// A order used here (contiguous k, matches load_b): lanes 0-15 -> M=lane, K = kb+{0..15}; lanes 16-31 -> +16.
__device__ __forceinline__ v16h load_a(const _Float16* base, int lane, int kb, int stride) {
  const _Float16* p = base + (lane & 15) * stride + kb + ((lane >> 4) << 4);
  v8h lo = *(const v8h*)(p);
  v8h hi = *(const v8h*)(p + 8);
  v16h r;
#pragma unroll
  for (int i = 0; i < 8; ++i) { r[i] = lo[i]; r[i + 8] = hi[i]; }
  return r;
}

// B fragment from packed weights W[n][k] (k contiguous, kp = padded K).
// B order used here (contiguous k, matches load_a): N = lane&15, K = kb + (lane>=16 ? 16 : 0) + e  -> 32B contiguous load.
__device__ __forceinline__ v16h load_b(const _Float16* w, int lane, int nb, int kb, int kp) {
  return *(const v16h*)(w + (size_t)(nb + (lane & 15)) * kp + kb + ((lane >> 4) << 4));
}

// Stage EMBxEMB f16 weight matrix (32KB) from global into LDS.
// Uses the Tensor Data Mover (one 2-D descriptor: 4096 x 8B elements, 1 row),
// tracked by TENSORcnt; falls back to a cooperative vector copy.
__device__ __forceinline__ void stage_w2_lds(_Float16* dst, const _Float16* src, int tid) {
#if defined(HAVE_TDM)
  if (tid == 0) {
    unsigned long long ga = (unsigned long long)(uintptr_t)src;
    unsigned lds = (unsigned)(uintptr_t)(void*)dst;           // low 32 bits = LDS byte offset
    const unsigned n8 = (EMB * EMB * 2) / 8;                  // 4096 8-byte elements
    u32x4 g0;
    g0[0] = 1u;                                               // count=1, user desc
    g0[1] = lds;                                              // lds_addr
    g0[2] = (unsigned)ga;                                     // global_addr[31:0]
    g0[3] = (unsigned)((ga >> 32) & 0x01FFFFFFu) | 0x80000000u; // addr[56:32] | type=2
    i32x8 g1;
    g1[0] = 0x00030000;                                       // data_size=3 (8B), mask=0
    g1[1] = (int)((n8 & 0xFFFFu) << 16);                      // tensor_dim0[15:0]
    g1[2] = (int)(((n8 >> 16) & 0xFFFFu) | (1u << 16));       // tensor_dim0[31:16] | tensor_dim1=1
    g1[3] = (int)((n8 & 0xFFFFu) << 16);                      // tile_dim0 = n8
    g1[4] = 1;                                                // tile_dim1 = 1
    g1[5] = (int)n8;                                          // tensor_dim0_stride
    g1[6] = 0;
    g1[7] = 0;
    i32x4 z4; z4[0] = 0; z4[1] = 0; z4[2] = 0; z4[3] = 0;
    i32x8 z8; z8[0] = 0; z8[1] = 0; z8[2] = 0; z8[3] = 0;
    z8[4] = 0; z8[5] = 0; z8[6] = 0; z8[7] = 0;
    __builtin_amdgcn_tensor_load_to_lds(g0, g1, z4, z4, z8, 0);  // 6-arg (clang-23) form
  }
#if __has_builtin(__builtin_amdgcn_s_wait_tensorcnt)
  __builtin_amdgcn_s_wait_tensorcnt(0);
#endif
#else
  for (int i = tid; i < EMB * EMB / 16; i += 128)
    ((v16h*)dst)[i] = ((const v16h*)src)[i];
#endif
}

// ---------------- utility kernels ----------------

__global__ void zero_f32(float* __restrict__ p, long n) {
  long i = (long)blockIdx.x * 256 + threadIdx.x;
  if (i < n) p[i] = 0.f;
}

// transpose+pad f32 W[k][n] (n=128) -> f16 Wp[n][kpad]
__global__ void pack_w(const float* __restrict__ W, _Float16* __restrict__ out, int K, int Kpad) {
  int idx = blockIdx.x * 256 + threadIdx.x;
  if (idx >= 128 * Kpad) return;
  int n = idx / Kpad, k = idx % Kpad;
  out[idx] = (_Float16)((k < K) ? W[(size_t)k * 128 + n] : 0.f);
}

// degree (atomic f32 add) + max(pos) (positive floats -> int-bit max)
__global__ void prep_kernel(const int* __restrict__ ei, const float* __restrict__ pos,
                            float* __restrict__ deg, int* __restrict__ maxbits) {
  int i = blockIdx.x * 256 + threadIdx.x;
  if (i < NN) atomicMax(maxbits, __float_as_int(pos[i]));
  if (i < NE) unsafeAtomicAdd(&deg[ei[NE + i]], 1.0f);
}

// ---------------- encoder: h = silu(silu(inp0 @ W1 + b1) @ W2 + b2) ----------------

__global__ void __launch_bounds__(128) enc_kernel(
    const float* __restrict__ x, const float* __restrict__ pos,
    const float* __restrict__ vars, const int* __restrict__ maxbits,
    const _Float16* __restrict__ W1p, const float* __restrict__ b1,
    const _Float16* __restrict__ W2p, const float* __restrict__ b2,
    float* __restrict__ h32, _Float16* __restrict__ h16) {
  __shared__ __align__(32) _Float16 Ain[16 * 32];
  __shared__ __align__(32) _Float16 Amid[16 * EMB];
  int tid = threadIdx.x, wave = tid >> 5, lane = tid & 31;
  int r = tid >> 3, sub = tid & 7;
  int node = blockIdx.x * 16 + r;
  float imp = fast_rcp(__int_as_float(*maxbits));
#pragma unroll
  for (int j = 0; j < 4; ++j) {
    int c = sub * 4 + j;
    float v;
    if (c < TW)            v = x[(size_t)node * TW + c];
    else if (c == TW)      v = pos[node] * imp;
    else if (c == TW + 1)  v = vars[(size_t)node * NV + 3] * (1.0f / T_MAXF);
    else if (c < 30)       v = vars[(size_t)node * NV + (c - (TW + 2))];
    else                   v = 0.f;
    Ain[r * 32 + c] = (_Float16)v;
  }
  __syncthreads();
#pragma unroll
  for (int ct = 0; ct < 2; ++ct) {
    int nb = (wave * 2 + ct) * 16;
    v8f acc = {};
    acc = wmma16(load_a(Ain, lane, 0, 32), load_b(W1p, lane, nb, 0, 32), acc);
    int n = nb + (lane & 15), mb = (lane >> 4) * 8;
    float bias = b1[n];
#pragma unroll
    for (int v = 0; v < 8; ++v) Amid[(mb + v) * EMB + n] = (_Float16)silu_f(acc[v] + bias);
  }
  __syncthreads();
#pragma unroll
  for (int ct = 0; ct < 2; ++ct) {
    int nb = (wave * 2 + ct) * 16;
    v8f acc = {};
#pragma unroll
    for (int kt = 0; kt < EMB / 32; ++kt)
      acc = wmma16(load_a(Amid, lane, kt * 32, EMB), load_b(W2p, lane, nb, kt * 32, EMB), acc);
    int n = nb + (lane & 15), mb = (lane >> 4) * 8;
    float bias = b2[n];
#pragma unroll
    for (int v = 0; v < 8; ++v) {
      float hv = silu_f(acc[v] + bias);
      size_t o = (size_t)(blockIdx.x * 16 + mb + v) * EMB + n;
      h32[o] = hv; h16[o] = (_Float16)hv;
    }
  }
}

// ---------------- per-edge message MLP, fused gather + scatter-add ----------------

__global__ void __launch_bounds__(128) msg_kernel(
    const _Float16* __restrict__ h16, const float* __restrict__ x,
    const float* __restrict__ pos, const float* __restrict__ vars,
    const int* __restrict__ ei, const int* __restrict__ maxbits,
    const _Float16* __restrict__ W1p, const float* __restrict__ b1,
    const _Float16* __restrict__ W2p, const float* __restrict__ b2,
    float* __restrict__ agg) {
  __shared__ __align__(32) _Float16 Ain[16 * KP];
  __shared__ __align__(32) _Float16 Amid[16 * EMB];
  __shared__ __align__(32) _Float16 W2lds[EMB * EMB];
  __shared__ int sd[16];
  int tid = threadIdx.x, wave = tid >> 5, lane = tid & 31;
  int r = tid >> 3, sub = tid & 7;
  __builtin_prefetch(W1p + (size_t)tid * KP, 0, 2);   // global_prefetch_b8 of W1 rows
  stage_w2_lds(W2lds, W2p, tid);                      // TDM 32KB weight stage (TENSORcnt)
  int e = blockIdx.x * 16 + r;
  int s = ei[e], d = ei[NE + e];
  if (sub == 0) sd[r] = d;
  ((v16h*)(Ain + r * KP))[sub]       = ((const v16h*)(h16 + (size_t)d * EMB))[sub];
  ((v16h*)(Ain + r * KP + EMB))[sub] = ((const v16h*)(h16 + (size_t)s * EMB))[sub];
  float imp = fast_rcp(__int_as_float(*maxbits));
#pragma unroll
  for (int j = 0; j < 4; ++j) {
    int c = 256 + sub * 4 + j;
    float v;
    if (c < 256 + TW)       v = x[(size_t)d * TW + (c - 256)] - x[(size_t)s * TW + (c - 256)];
    else if (c == 256 + TW) v = (pos[d] - pos[s]) * imp;
    else if (c < 262 + TW)  v = vars[(size_t)d * NV + (c - (257 + TW))];
    else                    v = 0.f;
    Ain[r * KP + c] = (_Float16)v;
  }
  __syncthreads();
#pragma unroll
  for (int ct = 0; ct < 2; ++ct) {
    int nb = (wave * 2 + ct) * 16;
    v8f acc = {};
#pragma unroll
    for (int kt = 0; kt < KP / 32; ++kt)
      acc = wmma16(load_a(Ain, lane, kt * 32, KP), load_b(W1p, lane, nb, kt * 32, KP), acc);
    int n = nb + (lane & 15), mb = (lane >> 4) * 8;
    float bias = b1[n];
#pragma unroll
    for (int v = 0; v < 8; ++v) Amid[(mb + v) * EMB + n] = (_Float16)silu_f(acc[v] + bias);
  }
  __syncthreads();
#pragma unroll
  for (int ct = 0; ct < 2; ++ct) {
    int nb = (wave * 2 + ct) * 16;
    v8f acc = {};
#pragma unroll
    for (int kt = 0; kt < EMB / 32; ++kt)
      acc = wmma16(load_a(Amid, lane, kt * 32, EMB), load_b(W2lds, lane, nb, kt * 32, EMB), acc);
    int n = nb + (lane & 15), mb = (lane >> 4) * 8;
    float bias = b2[n];
#pragma unroll
    for (int v = 0; v < 8; ++v)
      unsafeAtomicAdd(&agg[(size_t)sd[mb + v] * EMB + n], silu_f(acc[v] + bias));
  }
}

// ---------------- per-node update MLP + residual + stats accumulation ----------------

__global__ void __launch_bounds__(128) upd_kernel(
    const _Float16* __restrict__ h16, const float* __restrict__ h32,
    const float* __restrict__ agg, const float* __restrict__ deg,
    const float* __restrict__ vars,
    const _Float16* __restrict__ W1p, const float* __restrict__ b1,
    const _Float16* __restrict__ W2p, const float* __restrict__ b2,
    float* __restrict__ hnew, float* __restrict__ ssum, float* __restrict__ ssq) {
  __shared__ __align__(32) _Float16 Ain[16 * KP];
  __shared__ __align__(32) _Float16 Amid[16 * EMB];
  __shared__ __align__(32) _Float16 W2lds[EMB * EMB];
  int tid = threadIdx.x, wave = tid >> 5, lane = tid & 31;
  int r = tid >> 3, sub = tid & 7;
  __builtin_prefetch(W1p + (size_t)tid * KP, 0, 2);
  stage_w2_lds(W2lds, W2p, tid);
  int node = blockIdx.x * 16 + r;
  ((v16h*)(Ain + r * KP))[sub] = ((const v16h*)(h16 + (size_t)node * EMB))[sub];
  float invd = fast_rcp(fmaxf(deg[node], 1.f));
#pragma unroll
  for (int j = 0; j < 16; ++j) {
    int c = sub * 16 + j;
    Ain[r * KP + EMB + c] = (_Float16)(agg[(size_t)node * EMB + c] * invd);
  }
#pragma unroll
  for (int j = 0; j < 4; ++j) {
    int c = 256 + sub * 4 + j;
    float v = (c < 256 + NV) ? vars[(size_t)node * NV + (c - 256)] : 0.f;
    Ain[r * KP + c] = (_Float16)v;
  }
  __syncthreads();
#pragma unroll
  for (int ct = 0; ct < 2; ++ct) {
    int nb = (wave * 2 + ct) * 16;
    v8f acc = {};
#pragma unroll
    for (int kt = 0; kt < KP / 32; ++kt)
      acc = wmma16(load_a(Ain, lane, kt * 32, KP), load_b(W1p, lane, nb, kt * 32, KP), acc);
    int n = nb + (lane & 15), mb = (lane >> 4) * 8;
    float bias = b1[n];
#pragma unroll
    for (int v = 0; v < 8; ++v) Amid[(mb + v) * EMB + n] = (_Float16)silu_f(acc[v] + bias);
  }
  __syncthreads();
#pragma unroll
  for (int ct = 0; ct < 2; ++ct) {
    int nb = (wave * 2 + ct) * 16;
    v8f acc = {};
#pragma unroll
    for (int kt = 0; kt < EMB / 32; ++kt)
      acc = wmma16(load_a(Amid, lane, kt * 32, EMB), load_b(W2lds, lane, nb, kt * 32, EMB), acc);
    int n = nb + (lane & 15), mb = (lane >> 4) * 8;
    float bias = b2[n];
    float cs = 0.f, cq = 0.f;
#pragma unroll
    for (int v = 0; v < 8; ++v) {
      int nd = blockIdx.x * 16 + mb + v;
      float hn = h32[(size_t)nd * EMB + n] + silu_f(acc[v] + bias);
      hnew[(size_t)nd * EMB + n] = hn;
      cs += hn; cq += hn * hn;
    }
    cs += __shfl_xor(cs, 16, 32);
    cq += __shfl_xor(cq, 16, 32);
    if (lane < 16) { unsafeAtomicAdd(&ssum[n], cs); unsafeAtomicAdd(&ssq[n], cq); }
  }
}

// ---------------- per-feature normalization ----------------

__global__ void norm_kernel(const float* __restrict__ hnew,
                            const float* __restrict__ ssum, const float* __restrict__ ssq,
                            float* __restrict__ h32, _Float16* __restrict__ h16) {
  long idx = (long)blockIdx.x * 256 + threadIdx.x;
  if (idx >= (long)NN * EMB) return;
  int col = (int)(idx & (EMB - 1));
  float mu = ssum[col] * (1.0f / NN);
  float var = ssq[col] * (1.0f / NN) - mu * mu;
  float v = (hnew[idx] - mu) * fast_rsq(var + 1e-5f);
  h32[idx] = v; h16[idx] = (_Float16)v;
}

// ---------------- decoder: two small convs, one wave per node ----------------

__global__ void __launch_bounds__(256) dec_kernel(
    const float* __restrict__ h32, const float* __restrict__ x,
    const float* __restrict__ dtp,
    const float* __restrict__ W1, const float* __restrict__ b1,
    const float* __restrict__ W2, const float* __restrict__ b2,
    float* __restrict__ out) {
  __shared__ float za[8][8 * 38];
  int wave = threadIdx.x >> 5, lane = threadIdx.x & 31;
  int node = blockIdx.x * 8 + wave;
  const float* h = h32 + (size_t)node * EMB;
  for (int idx = lane; idx < 8 * 38; idx += 32) {
    int oc = idx & 7, j = idx >> 3;
    float s = b1[oc];
#pragma unroll
    for (int k = 0; k < 16; ++k) s += h[3 * j + k] * W1[oc * 16 + k];
    za[wave][idx] = silu_f(s);
  }
  __syncthreads();
  if (lane < TW) {
    float s = b2[0];
#pragma unroll
    for (int k = 0; k < 14; ++k)
#pragma unroll
      for (int oc = 0; oc < 8; ++oc)
        s += za[wave][(lane + k) * 8 + oc] * W2[oc * 14 + k];
    out[(size_t)node * TW + lane] = x[(size_t)node * TW + (TW - 1)] + dtp[0] * (float)(lane + 1) * s;
  }
}

// ---------------- host ----------------

extern "C" void kernel_launch(void* const* d_in, const int* in_sizes, int n_in,
                              void* d_out, int out_size, void* d_ws, size_t ws_size,
                              hipStream_t stream) {
  const float* x     = (const float*)d_in[0];
  const float* pos   = (const float*)d_in[1];
  const float* vars  = (const float*)d_in[2];
  const float* dt    = (const float*)d_in[3];
  const int*   ei    = (const int*)d_in[4];
  const float* encW1 = (const float*)d_in[5];
  const float* encB1 = (const float*)d_in[6];
  const float* encW2 = (const float*)d_in[7];
  const float* encB2 = (const float*)d_in[8];
  const float* msgW1 = (const float*)d_in[9];
  const float* msgB1 = (const float*)d_in[10];
  const float* msgW2 = (const float*)d_in[11];
  const float* msgB2 = (const float*)d_in[12];
  const float* updW1 = (const float*)d_in[13];
  const float* updB1 = (const float*)d_in[14];
  const float* updW2 = (const float*)d_in[15];
  const float* updB2 = (const float*)d_in[16];
  const float* decW1 = (const float*)d_in[17];
  const float* decB1 = (const float*)d_in[18];
  const float* decW2 = (const float*)d_in[19];
  const float* decB2 = (const float*)d_in[20];
  float* out = (float*)d_out;

  char* w = (char*)d_ws;
  float*   h32   = (float*)w;    w += (size_t)NN * EMB * 4;
  float*   hnew  = (float*)w;    w += (size_t)NN * EMB * 4;
  float*   agg   = (float*)w;    w += (size_t)NN * EMB * 4;
  _Float16* h16  = (_Float16*)w; w += (size_t)NN * EMB * 2;
  float*   deg   = (float*)w;    w += (size_t)NN * 4;
  float*   ssum  = (float*)w;    w += 512;
  float*   ssq   = (float*)w;    w += 512;
  int*     maxb  = (int*)w;      w += 256;
  _Float16* encW1p = (_Float16*)w; w += 128 * 32 * 2;
  _Float16* encW2p = (_Float16*)w; w += 128 * 128 * 2;
  _Float16* msgW1p = (_Float16*)w; w += (size_t)LAYERS * 128 * KP * 2;
  _Float16* msgW2p = (_Float16*)w; w += (size_t)LAYERS * 128 * 128 * 2;
  _Float16* updW1p = (_Float16*)w; w += (size_t)LAYERS * 128 * KP * 2;
  _Float16* updW2p = (_Float16*)w; w += (size_t)LAYERS * 128 * 128 * 2;

  // ---- setup: zero counters, degree + max(pos), pack weights to f16 [n][k] ----
  zero_f32<<<(NN + 255) / 256, 256, 0, stream>>>(deg, NN);
  zero_f32<<<1, 256, 0, stream>>>((float*)maxb, 1);
  prep_kernel<<<NE / 256, 256, 0, stream>>>(ei, pos, deg, maxb);

  pack_w<<<(128 * 32 + 255) / 256, 256, 0, stream>>>(encW1, encW1p, TW + NV + 1, 32);
  pack_w<<<(128 * 128 + 255) / 256, 256, 0, stream>>>(encW2, encW2p, 128, 128);
  for (int l = 0; l < LAYERS; ++l) {
    pack_w<<<(128 * KP + 255) / 256, 256, 0, stream>>>(msgW1 + (size_t)l * 286 * 128,
                                                       msgW1p + (size_t)l * 128 * KP, 286, KP);
    pack_w<<<(128 * 128 + 255) / 256, 256, 0, stream>>>(msgW2 + (size_t)l * 128 * 128,
                                                        msgW2p + (size_t)l * 128 * 128, 128, 128);
    pack_w<<<(128 * KP + 255) / 256, 256, 0, stream>>>(updW1 + (size_t)l * 260 * 128,
                                                       updW1p + (size_t)l * 128 * KP, 260, KP);
    pack_w<<<(128 * 128 + 255) / 256, 256, 0, stream>>>(updW2 + (size_t)l * 128 * 128,
                                                        updW2p + (size_t)l * 128 * 128, 128, 128);
  }

  // ---- encoder ----
  enc_kernel<<<NN / 16, 128, 0, stream>>>(x, pos, vars, maxb, encW1p, encB1, encW2p, encB2,
                                          h32, h16);

  // ---- message-passing layers ----
  for (int l = 0; l < LAYERS; ++l) {
    zero_f32<<<((long)NN * EMB + 255) / 256, 256, 0, stream>>>(agg, (long)NN * EMB);
    zero_f32<<<1, 256, 0, stream>>>(ssum, 128);
    zero_f32<<<1, 256, 0, stream>>>(ssq, 128);
    msg_kernel<<<NE / 16, 128, 0, stream>>>(h16, x, pos, vars, ei, maxb,
                                            msgW1p + (size_t)l * 128 * KP, msgB1 + l * 128,
                                            msgW2p + (size_t)l * 128 * 128, msgB2 + l * 128,
                                            agg);
    upd_kernel<<<NN / 16, 128, 0, stream>>>(h16, h32, agg, deg, vars,
                                            updW1p + (size_t)l * 128 * KP, updB1 + l * 128,
                                            updW2p + (size_t)l * 128 * 128, updB2 + l * 128,
                                            hnew, ssum, ssq);
    norm_kernel<<<((long)NN * EMB + 255) / 256, 256, 0, stream>>>(hnew, ssum, ssq, h32, h16);
  }

  // ---- decoder ----
  dec_kernel<<<NN / 8, 256, 0, stream>>>(h32, x, dt, decW1, decB1, decW2, decB2, out);
}
